// APAN_50251117363835
// MI455X (gfx1250) — matrix operand order repaired and checked by reading, not compiled, not run on hardware
//
#include <hip/hip_runtime.h>
#include <hip/hip_bf16.h>

// ---------------------------------------------------------------------------
// Problem constants (from reference)
// ---------------------------------------------------------------------------
#define BB        4096
#define NN        (3 * BB)          // 12288
#define D_MEM     128
#define D_EDGE    128
#define D_TIME    100
#define MAILS     10
#define D_MSG     (2 * D_MEM + D_EDGE)   // 384
#define D_KIN     (D_MSG + D_TIME)       // 484
#define KPAD      512                    // 484 padded to multiple of 32
#define EE        81920
#define NUNIQ     50000
#define LN_EPS    1e-5f

#define TILE_NODES 8                      // nodes per block in apan_node
#define MROWS      (TILE_NODES * MAILS)   // 80 mail rows per block

typedef __attribute__((ext_vector_type(16))) __bf16 v16bf;
typedef __attribute__((ext_vector_type(8)))  float  v8f;

// ---------------------------------------------------------------------------
// WMMA helpers (layouts per CDNA5 ISA sec 7.12.2, wave32)
// ---------------------------------------------------------------------------
__device__ inline v8f wmma_bf16(v16bf a, v16bf b, v8f c) {
  // 8 args: (neg_a, A, neg_b, B, c_mod, C, reuse_a, reuse_b)
  return __builtin_amdgcn_wmma_f32_16x16x32_bf16(false, a, false, b,
                                                 (short)0, c, false, false);
}

// A fragment from LDS: 16x32 bf16 tile, row-major leading dim `ld`.
// lane L: M = L&15; elements j=0..15 map to two contiguous 8-elem runs:
//   k = k0 + half*16 + j
// -> two 16-byte contiguous LDS reads per fragment (ds_load_b128 x2).
__device__ inline v16bf frag_a(const __bf16* src, int ld, int rbase, int k0, int lane) {
  const int m = lane & 15;
  const int half = lane >> 4;
  const __bf16* row = src + (long)(rbase + m) * ld + k0 + (half << 4);
  v16bf a;
#pragma unroll
  for (int j = 0; j < 8; ++j) a[j] = row[j];
#pragma unroll
  for (int j = 0; j < 8; ++j) a[8 + j] = row[8 + j];
  return a;
}

// B fragment from a fragment-ready packed weight buffer: each (ks, nt, lane)
// owns 16 contiguous bf16 (32 bytes) -> one vector load, no packing VALU.
__device__ inline v16bf frag_b_pk(const __bf16* Wf, int ntilesN, int ks, int nt, int lane) {
  const __bf16* p = Wf + (((long)(ks * ntilesN + nt) * 32 + lane) << 4);
  return *(const v16bf*)p;
}

// ---------------------------------------------------------------------------
// Weight pre-swizzle into fragment-ready layout (bf16).
// Fragment element j of (ks, nt, lane) = W[ks*32 + (lane>>4)*16 + j][nt*16 + (lane&15)]
// ---------------------------------------------------------------------------
__device__ inline void pack_w128(const float* __restrict__ W, __bf16* __restrict__ out, int i) {
  // W: 128x128 row-major [k][n]; i in [0, 4*8*32)
  const int lane = i & 31;
  const int nt   = (i >> 5) & 7;
  const int ks   = (i >> 5) >> 3;
  const int n  = nt * 16 + (lane & 15);
  const int kb = ks * 32 + ((lane >> 4) << 4);
  __bf16* dst = out + ((long)i << 4);
#pragma unroll
  for (int j = 0; j < 16; ++j) dst[j] = (__bf16)W[(kb + j) * D_MEM + n];
}

__device__ inline void pack_wkv(const float* __restrict__ wk, const float* __restrict__ wv,
                                __bf16* __restrict__ out, int i) {
  // virtual [512][256] = [wk | wv], K zero-padded 484 -> 512 ; i in [0, 16*16*32)
  const int lane = i & 31;
  const int nt   = (i >> 5) & 15;
  const int ks   = (i >> 5) >> 4;
  const int n  = nt * 16 + (lane & 15);
  const int kb = ks * 32 + ((lane >> 4) << 4);
  __bf16* dst = out + ((long)i << 4);
#pragma unroll
  for (int j = 0; j < 16; ++j) {
    int k = kb + j;
    float v = 0.f;
    if (k < D_KIN) v = (n < D_MEM) ? wk[k * D_MEM + n] : wv[k * D_MEM + (n - D_MEM)];
    dst[j] = (__bf16)v;
  }
}

__global__ void apan_prep(const float* __restrict__ wq, const float* __restrict__ wk,
                          const float* __restrict__ wv, const float* __restrict__ mlp,
                          const float* __restrict__ srcw, const float* __restrict__ dstw,
                          __bf16* wq16, __bf16* wkv16, __bf16* mlp16,
                          __bf16* src16, __bf16* dst16) {
  int i = blockIdx.x * 256 + threadIdx.x;
  if (i < 1024)       pack_w128(wq,   wq16,  i);
  else if (i < 2048)  pack_w128(mlp,  mlp16, i - 1024);
  else if (i < 3072)  pack_w128(srcw, src16, i - 2048);
  else if (i < 4096)  pack_w128(dstw, dst16, i - 3072);
  else if (i < 12288) pack_wkv(wk, wv, wkv16, i - 4096);
}

__global__ void apan_zero(float* p, long n) {
  long i = (long)blockIdx.x * blockDim.x + threadIdx.x;
  long stride = (long)gridDim.x * blockDim.x;
  for (; i < n; i += stride) p[i] = 0.f;
}

// pack two floats into one 32-bit bf16 pair store
__device__ inline void store_bf16x2(__bf16* base, float v0, float v1) {
  union { __bf16 h[2]; unsigned u; } pk;
  pk.h[0] = (__bf16)v0; pk.h[1] = (__bf16)v1;
  *(unsigned*)base = pk.u;
}

// ---------------------------------------------------------------------------
// Fused per-node attention block: gather + Q/K/V WMMA + softmax-attention +
// residual + LayerNorm + MLP. 8 nodes per block, 8 waves, dynamic LDS.
// ---------------------------------------------------------------------------
__global__ void __launch_bounds__(256)
apan_node(const int* __restrict__ nodes, const float* __restrict__ times,
          const float* __restrict__ mem_table, const float* __restrict__ mail_table,
          const float* __restrict__ mail_time,
          const float* __restrict__ te_w, const float* __restrict__ te_b,
          const float* __restrict__ wq_b, const float* __restrict__ wk_b,
          const float* __restrict__ wv_b, const float* __restrict__ ln_g,
          const float* __restrict__ ln_b, const float* __restrict__ mlp_b,
          const __bf16* __restrict__ wq16, const __bf16* __restrict__ wkv16,
          const __bf16* __restrict__ mlp16,
          float* __restrict__ mem_out) {
  extern __shared__ char smem[];
  __bf16* s_mail  = (__bf16*)(smem);            // 80*512*2  = 81920
  float*  s_kv    = (float*) (smem + 81920);    // 80*256*4  = 81920 (K|V)
  __bf16* s_mem16 = (__bf16*)(smem + 163840);   // 16*128*2  = 4096
  float*  s_memf  = (float*) (smem + 167936);   // 8*128*4   = 4096
  float*  s_q     = (float*) (smem + 172032);   // 16*128*4  = 8192
  float*  s_scr   = (float*) (smem + 180224);   // 8*2*10*4  = 640
  float*  s_out   = (float*) (smem + 180864);   // 8*128*4   = 4096
  __bf16* s_ln16  = (__bf16*)(smem + 184960);   // 16*128*2  = 4096
  // total 189056 B (< 320 KB/WGP)

  const int tid  = threadIdx.x;
  const int lane = tid & 31;
  const int wv   = tid >> 5;            // wave id 0..7
  const int n0   = blockIdx.x * TILE_NODES;

  // ---- Stage 1: gather memory rows (f32 + packed bf16), zero pad rows 8..15 ----
  for (int i = tid; i < TILE_NODES * D_MEM / 2; i += 256) {
    int r = i >> 6, dp = (i & 63) * 2;
    const float* row = mem_table + (long)nodes[n0 + r] * D_MEM + dp;
    float v0 = row[0], v1 = row[1];
    s_memf[r * D_MEM + dp]     = v0;
    s_memf[r * D_MEM + dp + 1] = v1;
    store_bf16x2(s_mem16 + r * D_MEM + dp, v0, v1);
    *(unsigned*)(s_mem16 + TILE_NODES * D_MEM + 2 * i) = 0u;   // pad rows 8..15
    *(unsigned*)(s_ln16  + TILE_NODES * D_MEM + 2 * i) = 0u;
  }
  __syncthreads();

  // ---- Stage 2: Q = mem @ wq + wq_b  (each wave owns one 16-col N tile) ----
  {
    v8f acc = {};
#pragma unroll
    for (int ks = 0; ks < 4; ++ks) {
      v16bf a = frag_a(s_mem16, D_MEM, 0, ks * 32, lane);
      v16bf b = frag_b_pk(wq16, 8, ks, wv, lane);
      acc = wmma_bf16(a, b, acc);
    }
    int col = wv * 16 + (lane & 15);
    int half = lane >> 4;
    float bias = wq_b[col];
#pragma unroll
    for (int v = 0; v < 8; ++v) {
      int row = v + half * 8;                 // rows 8..15 unused scratch
      s_q[row * D_MEM + col] = acc[v] + bias;
    }
  }

  // ---- Stage 3: build mail tile in LDS: [gathered 384 | cos time-enc | 0 pad] ----
  // pair-packed bf16 stores (regions start at even dp: no pair straddles)
  for (int i = tid; i < MROWS * (KPAD / 2); i += 256) {
    int r = i >> 8;                  // KPAD/2 == 256
    int dp = (i & 255) * 2;
    int nr = r / MAILS, m = r - nr * MAILS;
    int nd = nodes[n0 + nr];
    float v0, v1;
    if (dp < D_MSG) {
      const float* mrow = mail_table + ((long)nd * MAILS + m) * D_MSG + dp;
      v0 = mrow[0]; v1 = mrow[1];
    } else if (dp < D_MSG + D_TIME) {
      int j = dp - D_MSG;            // even, j+1 <= 99
      float dt = times[n0 + nr] - mail_time[(long)nd * MAILS + m];
      v0 = cosf(dt * te_w[j] + te_b[j]);
      v1 = cosf(dt * te_w[j + 1] + te_b[j + 1]);
    } else {
      v0 = 0.f; v1 = 0.f;
    }
    store_bf16x2(s_mail + r * KPAD + dp, v0, v1);
  }
  __syncthreads();

  // ---- Stage 4: [K|V] = mail @ [wk|wv] ----
  // wave owns N-tiles {wv, wv+8}; hoist one B fragment per k-step and feed
  // 5 M-tile accumulators (B traffic / 5, A streams from LDS as ds_load_b128).
  for (int hnt = 0; hnt < 2; ++hnt) {
    const int nt = wv + hnt * 8;
    v8f acc[5] = {};
#pragma unroll 1
    for (int ks = 0; ks < KPAD / 32; ++ks) {
      v16bf b = frag_b_pk(wkv16, 16, ks, nt, lane);
#pragma unroll
      for (int mt = 0; mt < 5; ++mt) {
        v16bf a = frag_a(s_mail, KPAD, mt * 16, ks * 32, lane);
        acc[mt] = wmma_bf16(a, b, acc[mt]);
      }
    }
    int col = nt * 16 + (lane & 15);
    int half = lane >> 4;
    float bias = (col < D_MEM) ? wk_b[col] : wv_b[col - D_MEM];
#pragma unroll
    for (int mt = 0; mt < 5; ++mt)
#pragma unroll
      for (int v = 0; v < 8; ++v) {
        int row = mt * 16 + v + half * 8;
        s_kv[row * 256 + col] = acc[mt][v] + bias;
      }
  }
  __syncthreads();

  // ---- Stage 5: attention scores (2 heads of 64), leaky relu ----
  if (tid < TILE_NODES * MAILS * 2) {
    int h = tid & 1, m = (tid >> 1) % MAILS, r = tid / (2 * MAILS);
    float s = 0.f;
    const float* qrow = s_q + r * D_MEM + h * 64;
    const float* krow = s_kv + (r * MAILS + m) * 256 + h * 64;
#pragma unroll 8
    for (int d = 0; d < 64; ++d) s += qrow[d] * krow[d];
    if (s < 0.f) s *= 0.2f;
    s_scr[(r * 2 + h) * MAILS + m] = s;
  }
  __syncthreads();

  // ---- Stage 6: softmax over MAIL dimension ----
  if (tid < TILE_NODES * 2) {
    float* sc = s_scr + tid * MAILS;
    float mx = sc[0];
#pragma unroll
    for (int m = 1; m < MAILS; ++m) mx = fmaxf(mx, sc[m]);
    float sum = 0.f;
#pragma unroll
    for (int m = 0; m < MAILS; ++m) { float e = expf(sc[m] - mx); sc[m] = e; sum += e; }
    float inv = 1.f / sum;
#pragma unroll
    for (int m = 0; m < MAILS; ++m) sc[m] *= inv;
  }
  __syncthreads();

  // ---- Stage 7: out = attn @ V + residual ----
  for (int i = tid; i < TILE_NODES * D_MEM; i += 256) {
    int r = i >> 7, d = i & 127, h = d >> 6;
    const float* aw = s_scr + (r * 2 + h) * MAILS;
    float o = 0.f;
#pragma unroll
    for (int m = 0; m < MAILS; ++m)
      o += aw[m] * s_kv[(r * MAILS + m) * 256 + D_MEM + d];
    s_out[i] = o + s_memf[i];
  }
  __syncthreads();

  // ---- Stage 8: LayerNorm, one wave per node; lane owns 4 contiguous dims ----
  {
    int r = wv;
    const float* xr = s_out + r * D_MEM + lane * 4;
    float vals[4], lsum = 0.f, lsq = 0.f;
#pragma unroll
    for (int j = 0; j < 4; ++j) {
      float x = xr[j];
      vals[j] = x; lsum += x; lsq += x * x;
    }
#pragma unroll
    for (int off = 16; off > 0; off >>= 1) {
      lsum += __shfl_xor(lsum, off, 32);
      lsq  += __shfl_xor(lsq,  off, 32);
    }
    float mu  = lsum * (1.f / D_MEM);
    float var = lsq * (1.f / D_MEM) - mu * mu;
    float rs  = rsqrtf(var + LN_EPS);
    float y[4];
#pragma unroll
    for (int j = 0; j < 4; ++j) {
      int d = lane * 4 + j;
      y[j] = (vals[j] - mu) * rs * ln_g[d] + ln_b[d];
    }
    store_bf16x2(s_ln16 + r * D_MEM + lane * 4,     y[0], y[1]);
    store_bf16x2(s_ln16 + r * D_MEM + lane * 4 + 2, y[2], y[3]);
  }
  __syncthreads();

  // ---- Stage 9: mem_out = relu(ln @ mlp_w + mlp_b) ----
  {
    v8f acc = {};
#pragma unroll
    for (int ks = 0; ks < 4; ++ks) {
      v16bf a = frag_a(s_ln16, D_MEM, 0, ks * 32, lane);
      v16bf b = frag_b_pk(mlp16, 8, ks, wv, lane);
      acc = wmma_bf16(a, b, acc);
    }
    int col = wv * 16 + (lane & 15);
    int half = lane >> 4;
    float bias = mlp_b[col];
#pragma unroll
    for (int v = 0; v < 8; ++v) {
      int row = v + half * 8;
      if (row < TILE_NODES)
        mem_out[(long)(n0 + row) * D_MEM + col] = fmaxf(acc[v] + bias, 0.f);
    }
  }
}

// ---------------------------------------------------------------------------
// Link predictor: pos/neg scores via 3 WMMA GEMMs + relu-dot with out_w
// ---------------------------------------------------------------------------
__global__ void __launch_bounds__(256)
apan_link(const float* __restrict__ mem_out,
          const float* __restrict__ src_b, const float* __restrict__ dst_b,
          const float* __restrict__ out_w, const float* __restrict__ out_b,
          const __bf16* __restrict__ src16, const __bf16* __restrict__ dst16,
          float* __restrict__ pos, float* __restrict__ negs) {
  __shared__ __bf16 s_s[16 * D_MEM], s_d[16 * D_MEM], s_n[16 * D_MEM];
  __shared__ float  s_hs[16 * D_MEM], s_hd[16 * D_MEM], s_hn[16 * D_MEM];
  __shared__ float  s_red[16 * 8 * 2];

  const int tid = threadIdx.x, lane = tid & 31, wv = tid >> 5;
  const int r0 = blockIdx.x * 16;

  for (int i = tid; i < 16 * D_MEM / 2; i += 256) {
    int r = i >> 6, dp = (i & 63) * 2;
    long o = (long)(r0 + r) * D_MEM + dp;
    store_bf16x2(s_s + r * D_MEM + dp, mem_out[o],               mem_out[o + 1]);
    store_bf16x2(s_d + r * D_MEM + dp, mem_out[o + (long)BB * D_MEM],
                                       mem_out[o + (long)BB * D_MEM + 1]);
    store_bf16x2(s_n + r * D_MEM + dp, mem_out[o + (long)2 * BB * D_MEM],
                                       mem_out[o + (long)2 * BB * D_MEM + 1]);
  }
  __syncthreads();

  {
    v8f a1 = {}, a2 = {}, a3 = {};
#pragma unroll
    for (int ks = 0; ks < 4; ++ks) {
      v16bf bs = frag_b_pk(src16, 8, ks, wv, lane);
      v16bf bd = frag_b_pk(dst16, 8, ks, wv, lane);
      v16bf as = frag_a(s_s, D_MEM, 0, ks * 32, lane);
      v16bf ad = frag_a(s_d, D_MEM, 0, ks * 32, lane);
      v16bf an = frag_a(s_n, D_MEM, 0, ks * 32, lane);
      a1 = wmma_bf16(as, bs, a1);
      a2 = wmma_bf16(ad, bd, a2);
      a3 = wmma_bf16(an, bd, a3);
    }
    int col = wv * 16 + (lane & 15);
    int half = lane >> 4;
    float bsv = src_b[col], bdv = dst_b[col];
#pragma unroll
    for (int v = 0; v < 8; ++v) {
      int row = v + half * 8;
      s_hs[row * D_MEM + col] = a1[v] + bsv;
      s_hd[row * D_MEM + col] = a2[v] + bdv;
      s_hn[row * D_MEM + col] = a3[v] + bdv;
    }
  }
  __syncthreads();

  if (tid < 128) {
    int r = tid >> 3, c = tid & 7;
    float pp = 0.f, pn = 0.f;
#pragma unroll
    for (int k = 0; k < 16; ++k) {
      int d = c * 16 + k;
      float hs = s_hs[r * D_MEM + d];
      float ow = out_w[d];
      pp += fmaxf(hs + s_hd[r * D_MEM + d], 0.f) * ow;
      pn += fmaxf(hs + s_hn[r * D_MEM + d], 0.f) * ow;
    }
    s_red[(r * 8 + c) * 2]     = pp;
    s_red[(r * 8 + c) * 2 + 1] = pn;
  }
  __syncthreads();

  if (tid < 16) {
    float pp = 0.f, pn = 0.f;
#pragma unroll
    for (int c = 0; c < 8; ++c) {
      pp += s_red[(tid * 8 + c) * 2];
      pn += s_red[(tid * 8 + c) * 2 + 1];
    }
    pos[r0 + tid]  = pp + out_b[0];
    negs[r0 + tid] = pn + out_b[0];
  }
}

// ---------------------------------------------------------------------------
// Segment scatter: one wave per edge, lanes stripe 384 dims; L2-resident atomics
// ---------------------------------------------------------------------------
__global__ void __launch_bounds__(256)
apan_scatter(const float* __restrict__ mem_out, const float* __restrict__ efeat,
             const float* __restrict__ times,
             const int* __restrict__ dstindex, const int* __restrict__ src_seg,
             float* __restrict__ mail_agg, float* __restrict__ ts_agg,
             float* __restrict__ cnt) {
  const int wv = threadIdx.x >> 5, lane = threadIdx.x & 31;
  const int e = blockIdx.x * 8 + wv;
  if (e >= EE) return;
  const int idx = dstindex[e];
  const int seg = src_seg[e];
  const int i1 = idx;
  const int i2 = (idx < BB) ? idx + BB : idx - BB;
  const int er = (idx < BB) ? idx : idx - BB;
  float* drow = mail_agg + (long)seg * D_MSG;
#pragma unroll
  for (int s = 0; s < D_MSG / 32; ++s) {
    int d = lane + s * 32;
    float v = (d < D_MEM)     ? mem_out[(long)i1 * D_MEM + d]
            : (d < 2 * D_MEM) ? mem_out[(long)i2 * D_MEM + (d - D_MEM)]
                              : efeat[(long)er * D_EDGE + (d - 2 * D_MEM)];
    atomicAdd(drow + d, v);
  }
  if (lane == 0) {
    atomicAdd(cnt + seg, 1.0f);
    atomicAdd(ts_agg + seg, times[idx]);
  }
}

__global__ void apan_final(float* __restrict__ mail_agg, float* __restrict__ ts_agg,
                           const float* __restrict__ cnt) {
  long i = (long)blockIdx.x * 256 + threadIdx.x;
  const long tot = (long)NUNIQ * D_MSG;
  if (i < tot) {
    int s = (int)(i / D_MSG);
    mail_agg[i] /= fmaxf(cnt[s], 1.f);
  } else if (i < tot + NUNIQ) {
    int s = (int)(i - tot);
    ts_agg[s] /= fmaxf(cnt[s], 1.f);
  }
}

// ---------------------------------------------------------------------------
// Launcher
// ---------------------------------------------------------------------------
extern "C" void kernel_launch(void* const* d_in, const int* in_sizes, int n_in,
                              void* d_out, int out_size, void* d_ws, size_t ws_size,
                              hipStream_t stream) {
  (void)in_sizes; (void)n_in; (void)out_size; (void)ws_size;

  const int*   nodes      = (const int*)  d_in[0];
  const float* times      = (const float*)d_in[1];
  const float* mem_table  = (const float*)d_in[2];
  const float* mail_table = (const float*)d_in[3];
  const float* mail_time  = (const float*)d_in[4];
  const float* efeat      = (const float*)d_in[5];
  const float* wq_w  = (const float*)d_in[6];
  const float* wq_b  = (const float*)d_in[7];
  const float* wk_w  = (const float*)d_in[8];
  const float* wk_b  = (const float*)d_in[9];
  const float* wv_w  = (const float*)d_in[10];
  const float* wv_b  = (const float*)d_in[11];
  const float* mlp_w = (const float*)d_in[12];
  const float* mlp_b = (const float*)d_in[13];
  const float* ln_g  = (const float*)d_in[14];
  const float* ln_b  = (const float*)d_in[15];
  const float* te_w  = (const float*)d_in[16];
  const float* te_b  = (const float*)d_in[17];
  const float* src_w = (const float*)d_in[18];
  const float* src_b = (const float*)d_in[19];
  const float* dst_w = (const float*)d_in[20];
  const float* dst_b = (const float*)d_in[21];
  const float* out_w = (const float*)d_in[22];
  const float* out_b = (const float*)d_in[23];
  const int* dstindex = (const int*)d_in[24];
  const int* src_seg  = (const int*)d_in[25];

  float* outf     = (float*)d_out;
  float* mem_out  = outf;                              // N*128
  float* pos      = mem_out + (long)NN * D_MEM;        // B
  float* negs     = pos + BB;                          // B
  float* mail_agg = negs + BB;                         // NUNIQ*384
  float* ts_agg   = mail_agg + (long)NUNIQ * D_MSG;    // NUNIQ

  char* ws = (char*)d_ws;
  __bf16* wq16  = (__bf16*)(ws);                 //  32768 B (frag-ready)
  __bf16* wkv16 = (__bf16*)(ws + 32768);         // 262144 B (frag-ready)
  __bf16* mlp16 = (__bf16*)(ws + 294912);        //  32768 B
  __bf16* src16 = (__bf16*)(ws + 327680);        //  32768 B
  __bf16* dst16 = (__bf16*)(ws + 360448);        //  32768 B
  float*  cnt   = (float*) (ws + 393216);        // 200000 B

  // 1. weight packing into fragment-ready bf16 layout
  apan_prep<<<48, 256, 0, stream>>>(wq_w, wk_w, wv_w, mlp_w, src_w, dst_w,
                                    wq16, wkv16, mlp16, src16, dst16);

  // 2. zero accumulation buffers (mail_agg + ts_agg contiguous, and cnt)
  apan_zero<<<2048, 256, 0, stream>>>(mail_agg, (long)NUNIQ * D_MSG + NUNIQ);
  apan_zero<<<256, 256, 0, stream>>>(cnt, (long)NUNIQ);

  // 3. fused node-attention block (dynamic LDS 189056 B)
  const size_t smem_node = 189056;
  (void)hipFuncSetAttribute(reinterpret_cast<const void*>(&apan_node),
                            hipFuncAttributeMaxDynamicSharedMemorySize,
                            (int)smem_node);
  apan_node<<<NN / TILE_NODES, 256, smem_node, stream>>>(
      nodes, times, mem_table, mail_table, mail_time, te_w, te_b,
      wq_b, wk_b, wv_b, ln_g, ln_b, mlp_b, wq16, wkv16, mlp16, mem_out);

  // 4. link prediction heads
  apan_link<<<BB / 16, 256, 0, stream>>>(mem_out, src_b, dst_b, out_w, out_b,
                                         src16, dst16, pos, negs);

  // 5. edge scatter (segment sums)
  apan_scatter<<<EE / 8, 256, 0, stream>>>(mem_out, efeat, times, dstindex,
                                           src_seg, mail_agg, ts_agg, cnt);

  // 6. normalize by counts
  apan_final<<<(int)(((long)NUNIQ * D_MSG + NUNIQ + 255) / 256), 256, 0, stream>>>(
      mail_agg, ts_agg, cnt);
}
